// DotProductAttention_8684423873252
// MI455X (gfx1250) — matrix operand order repaired and checked by reading, not compiled, not run on hardware
//
#include <hip/hip_runtime.h>

// Problem constants (from reference)
constexpr int Bc = 32;
constexpr int Qc = 1024;
constexpr int Kc = 1024;
constexpr int Dc = 512;
constexpr float MASKV = -1000000.0f;

typedef __bf16 bf16_t;
typedef __attribute__((ext_vector_type(16))) __bf16 v16bf;
typedef __attribute__((ext_vector_type(8)))  __bf16 v8bf;
typedef __attribute__((ext_vector_type(4)))  __bf16 v4bf;
typedef __attribute__((ext_vector_type(8)))  float  v8f;
typedef __attribute__((ext_vector_type(4)))  unsigned int ui32x4;
typedef __attribute__((ext_vector_type(8)))  int i32x8;
typedef __attribute__((ext_vector_type(4)))  int i32x4;

#define HAVE_TDM (__has_builtin(__builtin_amdgcn_tensor_load_to_lds) && \
                  __has_builtin(__builtin_amdgcn_s_wait_tensorcnt))

__device__ inline v8f wmma_bf16(v16bf a, v16bf b, v8f c) {
  return __builtin_amdgcn_wmma_f32_16x16x32_bf16(
      /*neg_a=*/false, a, /*neg_b=*/false, b,
      /*c_mod=*/(short)0, c, /*reuse_a=*/false, /*reuse_b=*/false);
}

__device__ inline v16bf cat8(v8bf lo, v8bf hi) {
  return __builtin_shufflevector(lo, hi, 0,1,2,3,4,5,6,7,8,9,10,11,12,13,14,15);
}

// ---------------- elementwise f32 -> bf16 convert ----------------
__global__ __launch_bounds__(256)
void cvt_bf16(const float* __restrict__ src, bf16_t* __restrict__ dst, int n4) {
  int i = blockIdx.x * blockDim.x + threadIdx.x;
  if (i >= n4) return;
  float4 v = reinterpret_cast<const float4*>(src)[i];
  v4bf o;
  o[0] = (bf16_t)v.x; o[1] = (bf16_t)v.y; o[2] = (bf16_t)v.z; o[3] = (bf16_t)v.w;
  reinterpret_cast<v4bf*>(dst)[i] = o;
}

// ------------- transpose convert: [b,k,d] f32 -> [b,d,k] bf16 -------------
__global__ __launch_bounds__(256)
void cvtT_bf16(const float* __restrict__ src, bf16_t* __restrict__ dst) {
  __shared__ float tile[32][33];
  const int b  = blockIdx.z;
  const int k0 = blockIdx.x * 32;
  const int d0 = blockIdx.y * 32;
  const int tx = threadIdx.x;   // 0..31
  const int ty = threadIdx.y;   // 0..7
  const float* s = src + ((size_t)b * Kc + k0) * Dc + d0;
  #pragma unroll
  for (int r = ty; r < 32; r += 8)
    tile[r][tx] = s[(size_t)r * Dc + tx];
  __syncthreads();
  bf16_t* o = dst + ((size_t)b * Dc + d0) * Kc + k0;
  #pragma unroll
  for (int r = ty; r < 32; r += 8)
    o[(size_t)r * Kc + tx] = (bf16_t)tile[tx][r];
}

// ---------------- proj = Q @ W^T  (bf16 WMMA, f32 acc, bf16 out) ----------------
__global__ __launch_bounds__(256)
void proj_gemm(const bf16_t* __restrict__ qb, const bf16_t* __restrict__ wb,
               bf16_t* __restrict__ pb) {
  const int lane = threadIdx.x & 31;
  const int wave = threadIdx.x >> 5;
  const int tile = blockIdx.x * 8 + wave;   // (B*Q/16) * (D/16) tiles
  const int mt = tile >> 5;                 // / (D/16 = 32)
  const int nt = tile & 31;
  const int ln16 = lane & 15;
  const int half = lane >> 4;

  const bf16_t* arow = qb + (size_t)(mt * 16 + ln16) * Dc;
  const bf16_t* brow = wb + (size_t)(nt * 16 + ln16) * Dc;
  v8f c = {};
  #pragma unroll 4
  for (int kt = 0; kt < Dc / 32; ++kt) {
    const int da = kt * 32 + half * 16;
    v8bf alo = *(const v8bf*)(arow + da);
    v8bf ahi = *(const v8bf*)(arow + da + 8);
    v16bf bm = *(const v16bf*)(brow + kt * 32 + half * 16);
    c = wmma_bf16(cat8(alo, ahi), bm, c);
  }
  // C layout: N = lane%16 ; M = r + 8*(lane>=16)
  bf16_t* prow = pb + (size_t)(mt * 16 + half * 8) * Dc + nt * 16 + ln16;
  #pragma unroll
  for (int r = 0; r < 8; ++r)
    prow[(size_t)r * Dc] = (bf16_t)c[r];
}

// ---------------- flash attention with dual (V, L) accumulation ----------------
// pb:[B*Q,D] bf16, kb:[B,K,D] bf16, vbT/lbT:[B,D,K] bf16
__global__ __launch_bounds__(256)
void flash_attn(const bf16_t* __restrict__ pb, const bf16_t* __restrict__ kb,
                const bf16_t* __restrict__ vbT, const bf16_t* __restrict__ lbT,
                const long long* __restrict__ valid_lens,
                float* __restrict__ outV, float* __restrict__ outL) {
  constexpr int QPAD = Dc + 8;   // 520 (matches TDM pad: 256 DW interval + 4 DW pad)
  constexpr int PPAD = 128 + 8;  // 136
  __shared__ bf16_t sProj[16][QPAD];
  __shared__ bf16_t sP[16][PPAD];
  __shared__ float  sRed[8][16];
  __shared__ float  sMnew[16];
  __shared__ float  sCorr[16];
  __shared__ float  sMrun[16];
  __shared__ float  sLrun[16];

  const int b    = blockIdx.y;
  const int q0   = blockIdx.x * 16;
  const int tid  = threadIdx.x;
  const int lane = tid & 31;
  const int wave = tid >> 5;
  const int ln16 = lane & 15;
  const int half = lane >> 4;

  // ---- stage proj tile (16 x 512 bf16) into LDS ----
  const bf16_t* psrc = pb + ((size_t)b * Qc + q0) * Dc;
#if HAVE_TDM
  if (tid == 0) {
    // Tensor DMA descriptor (D#): 2D tile 512x16 of 2-byte elems, LDS padding
    // of 4 DWORDs every 256 DWORDs -> 520-element row stride in LDS.
    ui32x4 g0;
    const unsigned long long ga = (unsigned long long)(size_t)psrc;
    g0[0] = 1u;                                       // count=1 (valid), user D#
    g0[1] = (unsigned)(size_t)(&sProj[0][0]);         // lds_addr (low 32 bits)
    g0[2] = (unsigned)ga;                             // global_addr[31:0]
    g0[3] = (unsigned)((ga >> 32) & 0x01FFFFFFull) | 0x80000000u; // addr[56:32]|type=2
    i32x8 g1;
    g1[0] = 0x07D10000;   // data_size=2B, pad_enable, pad_interval=7(256DW), pad_amount=3(4DW)
    g1[1] = 0x02000000;   // tensor_dim0 = 512 (bits 63:48)
    g1[2] = 0x00100000;   // tensor_dim1 = 16  (bits 95:80)
    g1[3] = 0x02000000;   // tile_dim0   = 512 (bits 127:112)
    g1[4] = 16;           // tile_dim1   = 16
    g1[5] = 512;          // tensor_dim0_stride = 512
    g1[6] = 0;
    g1[7] = 0;
    const i32x4 z4 = {0, 0, 0, 0};                    // 2D tensor: groups 2/3 unused
    const i32x8 z8 = {0, 0, 0, 0, 0, 0, 0, 0};        // extra group (clang-23 form)
    __builtin_amdgcn_tensor_load_to_lds(g0, g1, z4, z4, z8, 0);
  }
  __builtin_amdgcn_s_wait_tensorcnt(0);
#else
  for (int chunk = tid; chunk < 16 * (Dc / 8); chunk += 256) {
    const int r  = chunk >> 6;            // / (Dc/8)
    const int c8 = (chunk & 63) * 8;
    *(v8bf*)&sProj[r][c8] = *(const v8bf*)(psrc + (size_t)r * Dc + c8);
  }
#endif
  if (tid < 16) { sMrun[tid] = -3.0e38f; sLrun[tid] = 0.0f; }
  __syncthreads();

  const int   vlen  = (int)valid_lens[b];
  const float scale = 0.044194173824159216f;  // 1/sqrt(512)

  v8f accV[4] = {};
  v8f accL[4] = {};
  const bf16_t* kbatch = kb + (size_t)b * Kc * Dc;
  const bf16_t* vtp[4];
  const bf16_t* ltp[4];
  #pragma unroll
  for (int nt = 0; nt < 4; ++nt) {
    const int col = wave * 64 + nt * 16 + ln16;
    vtp[nt] = vbT + ((size_t)b * Dc + col) * Kc;
    ltp[nt] = lbT + ((size_t)b * Dc + col) * Kc;
  }

  for (int blk = 0; blk < Kc / 128; ++blk) {
    const int kblk  = blk * 128;
    const int mykey = kblk + wave * 16 + ln16;   // this lane's key column

    // prefetch next block's operands into cache while we compute this one
    if (blk + 1 < Kc / 128) {
      __builtin_prefetch(kbatch + (size_t)(mykey + 128) * Dc, 0, 1);
      __builtin_prefetch(vtp[0] + kblk + 128, 0, 1);
      __builtin_prefetch(ltp[0] + kblk + 128, 0, 1);
    }

    // ---- S = proj tile (16xD) x K_block^T : this wave's 16x16 score tile ----
    v8f s = {};
    const bf16_t* krow = kbatch + (size_t)mykey * Dc;
    #pragma unroll 4
    for (int kt = 0; kt < Dc / 32; ++kt) {
      const int da = kt * 32 + half * 16;
      v8bf alo = *(const v8bf*)&sProj[ln16][da];
      v8bf ahi = *(const v8bf*)&sProj[ln16][da + 8];
      v16bf bm = *(const v16bf*)(krow + kt * 32 + half * 16);
      s = wmma_bf16(cat8(alo, ahi), bm, s);
    }

    // scale + valid_len mask (all 8 elems of a lane share one key index)
    const bool masked = (mykey >= vlen);
    float mloc[8];
    #pragma unroll
    for (int r = 0; r < 8; ++r) {
      const float v = masked ? MASKV : s[r] * scale;
      s[r] = v;
      mloc[r] = v;
    }
    // row max across 16 key-columns (each 16-lane half independently)
    #pragma unroll
    for (int off = 1; off < 16; off <<= 1) {
      #pragma unroll
      for (int r = 0; r < 8; ++r)
        mloc[r] = fmaxf(mloc[r], __shfl_xor(mloc[r], off, 16));
    }
    if (ln16 == 0) {
      #pragma unroll
      for (int r = 0; r < 8; ++r) sRed[wave][half * 8 + r] = mloc[r];
    }
    __syncthreads();

    // online-softmax stats update (one thread per row)
    if (tid < 16) {
      float mb = sRed[0][tid];
      #pragma unroll
      for (int w = 1; w < 8; ++w) mb = fmaxf(mb, sRed[w][tid]);
      const float mold = sMrun[tid];
      const float mnew = fmaxf(mold, mb);
      sMnew[tid] = mnew;
      sCorr[tid] = __expf(mold - mnew);
      sMrun[tid] = mnew;
    }
    __syncthreads();

    float rowm[8], corr8[8];
    #pragma unroll
    for (int r = 0; r < 8; ++r) {
      rowm[r]  = sMnew[half * 8 + r];
      corr8[r] = sCorr[half * 8 + r];
    }
    // p = exp(s - m_new); stash bf16 P tile in LDS; accumulate local row sums
    float psum[8];
    #pragma unroll
    for (int r = 0; r < 8; ++r) {
      const float p = __expf(s[r] - rowm[r]);
      psum[r] = p;
      sP[half * 8 + r][wave * 16 + ln16] = (bf16_t)p;
    }
    #pragma unroll
    for (int off = 1; off < 16; off <<= 1) {
      #pragma unroll
      for (int r = 0; r < 8; ++r)
        psum[r] += __shfl_xor(psum[r], off, 16);
    }
    if (ln16 == 0) {
      #pragma unroll
      for (int r = 0; r < 8; ++r) sRed[wave][half * 8 + r] = psum[r];
    }
    // rescale running accumulators by exp(m_old - m_new)
    #pragma unroll
    for (int nt = 0; nt < 4; ++nt) {
      #pragma unroll
      for (int r = 0; r < 8; ++r) {
        accV[nt][r] *= corr8[r];
        accL[nt][r] *= corr8[r];
      }
    }
    __syncthreads();

    if (tid < 16) {
      float ps = 0.0f;
      #pragma unroll
      for (int w = 0; w < 8; ++w) ps += sRed[w][tid];
      sLrun[tid] = sLrun[tid] * sCorr[tid] + ps;
    }

    // ---- O_v += P x V_block, O_l += P x L_block (this wave's 64 columns) ----
    #pragma unroll
    for (int kt = 0; kt < 4; ++kt) {
      const int da = kt * 32 + half * 16;
      v8bf plo = *(const v8bf*)&sP[ln16][da];
      v8bf phi = *(const v8bf*)&sP[ln16][da + 8];
      const v16bf a = cat8(plo, phi);
      const int kk = kblk + kt * 32 + half * 16;
      #pragma unroll
      for (int nt = 0; nt < 4; ++nt) {
        v16bf bv = *(const v16bf*)(vtp[nt] + kk);
        accV[nt] = wmma_bf16(a, bv, accV[nt]);
        v16bf bl = *(const v16bf*)(ltp[nt] + kk);
        accL[nt] = wmma_bf16(a, bl, accL[nt]);
      }
    }
    __syncthreads();
  }

  // final 1/l normalization and store
  if (tid < 16) sMnew[tid] = 1.0f / sLrun[tid];
  __syncthreads();
  float inv8[8];
  #pragma unroll
  for (int r = 0; r < 8; ++r) inv8[r] = sMnew[half * 8 + r];

  const size_t obase = ((size_t)b * Qc + q0 + half * 8) * Dc;
  #pragma unroll
  for (int nt = 0; nt < 4; ++nt) {
    const int col = wave * 64 + nt * 16 + ln16;
    #pragma unroll
    for (int r = 0; r < 8; ++r) {
      const size_t idx = obase + (size_t)r * Dc + col;
      outV[idx] = accV[nt][r] * inv8[r];
      outL[idx] = accL[nt][r] * inv8[r];
    }
  }
}

extern "C" void kernel_launch(void* const* d_in, const int* in_sizes, int n_in,
                              void* d_out, int out_size, void* d_ws, size_t ws_size,
                              hipStream_t stream) {
  (void)in_sizes; (void)n_in; (void)out_size; (void)ws_size;
  const float*     q  = (const float*)d_in[0];
  const float*     k  = (const float*)d_in[1];
  const float*     v  = (const float*)d_in[2];
  const float*     l  = (const float*)d_in[3];
  const float*     W  = (const float*)d_in[4];
  const long long* vl = (const long long*)d_in[5];   // reference: int64

  const size_t NQ = (size_t)Bc * Qc * Dc;  // 16.7M elems
  const size_t NK = (size_t)Bc * Kc * Dc;
  const size_t NW = (size_t)Dc * Dc;

  bf16_t* qb  = (bf16_t*)d_ws;
  bf16_t* kb  = qb  + NQ;
  bf16_t* vbT = kb  + NK;
  bf16_t* lbT = vbT + NK;
  bf16_t* wb  = lbT + NK;
  bf16_t* pb  = wb  + NW;

  cvt_bf16<<<(int)(NQ / 4 + 255) / 256, 256, 0, stream>>>(q, qb, (int)(NQ / 4));
  cvt_bf16<<<(int)(NK / 4 + 255) / 256, 256, 0, stream>>>(k, kb, (int)(NK / 4));
  cvt_bf16<<<(int)(NW / 4 + 255) / 256, 256, 0, stream>>>(W, wb, (int)(NW / 4));

  dim3 tg(Kc / 32, Dc / 32, Bc);
  cvtT_bf16<<<tg, dim3(32, 8, 1), 0, stream>>>(v, vbT);
  cvtT_bf16<<<tg, dim3(32, 8, 1), 0, stream>>>(l, lbT);

  proj_gemm<<<(Bc * Qc / 16) * (Dc / 16) / 8, 256, 0, stream>>>(qb, wb, pb);

  float* outV = (float*)d_out;
  float* outL = outV + NQ;
  flash_attn<<<dim3(Qc / 16, Bc), 256, 0, stream>>>(pb, kb, vbT, lbT, vl, outV, outL);
}
